// AttentionBasedMultiModalFusion_9998683865389
// MI455X (gfx1250) — compile-verified, matrix operand order checked by reading, not run on hardware
//
#include <hip/hip_runtime.h>
#include <stdint.h>

typedef unsigned short u16;
typedef __attribute__((ext_vector_type(16))) __bf16 v16bf;
typedef __attribute__((ext_vector_type(8)))  float  v8f;
typedef __attribute__((ext_vector_type(4)))  float  f32x4;

// ---------------------------------------------------------------- helpers
__device__ __forceinline__ float bf2f(u16 u) {
  unsigned x = ((unsigned)u) << 16;
  return __builtin_bit_cast(float, x);
}
__device__ __forceinline__ u16 f2bf(float f) {  // RNE fp32 -> bf16
  unsigned u = __builtin_bit_cast(unsigned, f);
  unsigned r = u + 0x7FFFu + ((u >> 16) & 1u);
  return (u16)(r >> 16);
}
__device__ __forceinline__ float sigmoidf(float x) { return 1.0f / (1.0f + __expf(-x)); }

__device__ __forceinline__ float wave_sum(float v) {
#pragma unroll
  for (int o = 16; o > 0; o >>= 1) v += __shfl_xor(v, o, 32);
  return v;
}
__device__ __forceinline__ float wave_max(float v) {
#pragma unroll
  for (int o = 16; o > 0; o >>= 1) v = fmaxf(v, __shfl_xor(v, o, 32));
  return v;
}
__device__ __forceinline__ float block_sum(float v, float* sm) {
  __syncthreads();
  int lane = threadIdx.x & 31, w = threadIdx.x >> 5, nw = blockDim.x >> 5;
  v = wave_sum(v);
  if (lane == 0) sm[w] = v;
  __syncthreads();
  float r = (threadIdx.x < (unsigned)nw) ? sm[threadIdx.x] : 0.f;
  if (w == 0) { r = wave_sum(r); if (lane == 0) sm[0] = r; }
  __syncthreads();
  return sm[0];
}
__device__ __forceinline__ float block_max(float v, float* sm) {
  __syncthreads();
  int lane = threadIdx.x & 31, w = threadIdx.x >> 5, nw = blockDim.x >> 5;
  v = wave_max(v);
  if (lane == 0) sm[w] = v;
  __syncthreads();
  float r = (threadIdx.x < (unsigned)nw) ? sm[threadIdx.x] : -__builtin_inff();
  if (w == 0) { r = wave_max(r); if (lane == 0) sm[0] = r; }
  __syncthreads();
  return sm[0];
}

// ---------------------------------------------------------------- fp32 -> bf16 (with K padding)
__global__ void cvt_pad(const float* __restrict__ src, u16* __restrict__ dst,
                        int rows, int K, int Kp) {
  int i = blockIdx.x * blockDim.x + threadIdx.x;
  int total = rows * Kp;
  if (i >= total) return;
  int r = i / Kp, k = i - r * Kp;
  dst[i] = (k < K) ? f2bf(src[(size_t)r * K + k]) : (u16)0;
}

__global__ void zero_f32(float* p, int n) {
  int i = blockIdx.x * blockDim.x + threadIdx.x;
  if (i < n) p[i] = 0.f;
}

// ---------------------------------------------------------------- WMMA bf16 GEMM: C[M,N] = A[M,K] * W[N,K]^T
// A fragment per ISA 7.12.2: lanes 0-15 hold M=0..15 with K=[sel*8..sel*8+8)+[16+sel*8..), sel=lane>>4
__device__ __forceinline__ v16bf ldfrag(const u16* q) {
  union { f32x4 c[2]; v16bf v; } u;
  u.c[0] = *(const f32x4*)q;         // K chunk 0 (16B)
  u.c[1] = *(const f32x4*)(q + 16);  // K chunk 1 (16B, +16 elements)
  return u.v;
}

__global__ __launch_bounds__(256) void wmma_gemm_bf16(const u16* __restrict__ A,
                                                      const u16* __restrict__ W,
                                                      float* __restrict__ C,
                                                      int M, int N, int K) {
  const int w = threadIdx.x >> 5, lane = threadIdx.x & 31;
  const int sel = lane >> 4, l15 = lane & 15;
  const int m0 = blockIdx.x * 64 + (w >> 2) * 32;   // 2 waves along M
  const int n0 = blockIdx.y * 128 + (w & 3) * 32;   // 4 waves along N
  int ra0 = m0 + l15;      if (ra0 > M - 1) ra0 = M - 1;
  int ra1 = m0 + 16 + l15; if (ra1 > M - 1) ra1 = M - 1;
  const u16* pa0 = A + (size_t)ra0 * K + sel * 8;
  const u16* pa1 = A + (size_t)ra1 * K + sel * 8;
  const u16* pb0 = W + (size_t)(n0 + l15) * K + sel * 8;
  const u16* pb1 = W + (size_t)(n0 + 16 + l15) * K + sel * 8;
  v8f c00{}, c01{}, c10{}, c11{};
  for (int k = 0; k < K; k += 32) {
    __builtin_prefetch(pa0 + 256, 0, 1);   // global_prefetch_b8, 8 k-steps ahead
    __builtin_prefetch(pb0 + 256, 0, 1);
    v16bf a0 = ldfrag(pa0), a1 = ldfrag(pa1);
    v16bf b0 = ldfrag(pb0), b1 = ldfrag(pb1);
    c00 = __builtin_amdgcn_wmma_f32_16x16x32_bf16(false, a0, false, b0, (short)0, c00, false, false);
    c01 = __builtin_amdgcn_wmma_f32_16x16x32_bf16(false, a0, false, b1, (short)0, c01, false, false);
    c10 = __builtin_amdgcn_wmma_f32_16x16x32_bf16(false, a1, false, b0, (short)0, c10, false, false);
    c11 = __builtin_amdgcn_wmma_f32_16x16x32_bf16(false, a1, false, b1, (short)0, c11, false, false);
    pa0 += 32; pa1 += 32; pb0 += 32; pb1 += 32;
  }
  // C/D layout: VGPR r -> M = r + 8*sel, N = lane&15
  const int col = n0 + l15;
#pragma unroll
  for (int r = 0; r < 8; r++) {
    int row = m0 + r + 8 * sel;
    if (row < M) { C[(size_t)row * N + col] = c00[r]; C[(size_t)row * N + col + 16] = c01[r]; }
    int row2 = row + 16;
    if (row2 < M) { C[(size_t)row2 * N + col] = c10[r]; C[(size_t)row2 * N + col + 16] = c11[r]; }
  }
}

// ---------------------------------------------------------------- persistent LSTM scan (1 workgroup / direction)
__device__ void lstm_scan_body(const float* __restrict__ xp, const u16* __restrict__ Wh,
                               const float* __restrict__ b, float* __restrict__ out,
                               int L, int H, int ldo, int colOff, bool rev,
                               float* h_s, float* g_s) {
  const int tid = threadIdx.x, nt = blockDim.x, H4 = 4 * H;
  float c = 0.f;
  if (tid < H) h_s[tid] = 0.f;
  __syncthreads();
  for (int t = 0; t < L; t++) {
    const int idx = rev ? (L - 1 - t) : t;
    const float* xr = xp + (size_t)idx * H4;
    for (int r = tid; r < H4; r += nt) {
      const u16* wr = Wh + (size_t)r * H;
      float acc = xr[r] + b[r];
#pragma unroll 8
      for (int k = 0; k < H; k++) acc += bf2f(wr[k]) * h_s[k];
      g_s[r] = acc;
    }
    __syncthreads();
    if (tid < H) {
      float ig = g_s[tid], fg = g_s[H + tid], gg = g_s[2 * H + tid], og = g_s[3 * H + tid];
      c = sigmoidf(fg) * c + sigmoidf(ig) * tanhf(gg);
      float h = sigmoidf(og) * tanhf(c);
      h_s[tid] = h;
      out[(size_t)idx * ldo + colOff + tid] = h;
    }
    __syncthreads();
  }
}

__global__ __launch_bounds__(1024) void lstm_scan4(
    const float* xpIF, const float* xpIB, const float* xpQF, const float* xpQB,
    const u16* WhIF, const u16* WhIB, const u16* WhQF, const u16* WhQB,
    const float* bIF, const float* bIB, const float* bQF, const float* bQB,
    float* img_emb, float* q_emb, int LI_, int LQ_) {
  __shared__ float h_s[512];
  __shared__ float g_s[2048];
  switch (blockIdx.x) {
    case 0:  lstm_scan_body(xpIF, WhIF, bIF, img_emb, LI_, 512, 1024, 0,   false, h_s, g_s); break;
    case 1:  lstm_scan_body(xpIB, WhIB, bIB, img_emb, LI_, 512, 1024, 512, true,  h_s, g_s); break;
    case 2:  lstm_scan_body(xpQF, WhQF, bQF, q_emb,  LQ_, 512, 1024, 0,   false, h_s, g_s); break;
    default: lstm_scan_body(xpQB, WhQB, bQB, q_emb,  LQ_, 512, 1024, 512, true,  h_s, g_s); break;
  }
}

// ---------------------------------------------------------------- wave-per-row matvec (fp32): out[r] = M[r]·v + bias
__global__ __launch_bounds__(256) void rowdot(const float* __restrict__ Mx, const float* __restrict__ v,
                                              const float* __restrict__ bias, int bias_scalar,
                                              float* __restrict__ out, int rows, int cols) {
  const int wid = (blockIdx.x * blockDim.x + threadIdx.x) >> 5;
  const int lane = threadIdx.x & 31;
  if (wid >= rows) return;
  const float* row = Mx + (size_t)wid * cols;
  float acc = 0.f;
  for (int k = lane; k < cols; k += 32) acc += row[k] * v[k];
  acc = wave_sum(acc);
  if (lane == 0) out[wid] = acc + (bias ? (bias_scalar ? bias[0] : bias[wid]) : 0.f);
}

// ---------------------------------------------------------------- attention: softmax(Watt·h + e) @ emb
__global__ __launch_bounds__(1024) void attn_ctx(const float* __restrict__ Watt, const float* __restrict__ h,
                                                 const float* __restrict__ e, const float* __restrict__ emb,
                                                 float* __restrict__ ctx, int rows, int cols, int hlen) {
  __shared__ float red[32];
  __shared__ float a_s[1024];
  const int tid = threadIdx.x;
  float p = 0.f;
  for (int k = tid; k < hlen; k += blockDim.x) p += Watt[k] * h[k];
  float hdot = block_sum(p, red);
  float s  = (tid < rows) ? (hdot + e[tid]) : -__builtin_inff();
  float m  = block_max(s, red);
  float ex = (tid < rows) ? __expf(s - m) : 0.f;
  float sum = block_sum(ex, red);
  if (tid < rows) a_s[tid] = ex / sum;
  __syncthreads();
  for (int j = tid; j < cols; j += blockDim.x) {
    float acc = 0.f;
    for (int i = 0; i < rows; i++) acc += a_s[i] * emb[(size_t)i * cols + j];
    ctx[j] = acc;
  }
}

// ---------------------------------------------------------------- modality weights mw[2]
__global__ __launch_bounds__(256) void mod_weights(const float* __restrict__ Wmod, const float* __restrict__ bmod,
                                                   const float* __restrict__ Wmi, const float* __restrict__ Wmq,
                                                   const float* __restrict__ h, const float* __restrict__ ci,
                                                   const float* __restrict__ cq, float* __restrict__ mw, int n) {
  __shared__ float red[32];
  float p0 = 0.f, p1 = 0.f, p2 = 0.f;
  for (int k = threadIdx.x; k < n; k += blockDim.x) {
    p0 += Wmod[k] * h[k]; p1 += Wmi[k] * ci[k]; p2 += Wmq[k] * cq[k];
  }
  float d0 = block_sum(p0, red), d1 = block_sum(p1, red), d2 = block_sum(p2, red);
  if (threadIdx.x == 0) {
    float tmp = d0 + bmod[0];
    float wi = tanhf(tmp + d1), wq = tanhf(tmp + d2);
    float m = fmaxf(wi, wq);
    float e0 = __expf(wi - m), e1 = __expf(wq - m);
    mw[0] = e0 / (e0 + e1); mw[1] = e1 / (e0 + e1);
  }
}

// ---------------------------------------------------------------- fs[r] = tanh(Wfus[r]·h + bfus + mw0*Wfi[r]·ci + mw1*Wfq[r]·cq)
__global__ __launch_bounds__(256) void fuse_fs(const float* __restrict__ Wfus, const float* __restrict__ bfus,
                                               const float* __restrict__ Wfi, const float* __restrict__ Wfq,
                                               const float* __restrict__ h, const float* __restrict__ ci,
                                               const float* __restrict__ cq, const float* __restrict__ mw,
                                               float* __restrict__ fs, int rows, int n) {
  const int wid = (blockIdx.x * blockDim.x + threadIdx.x) >> 5;
  const int lane = threadIdx.x & 31;
  if (wid >= rows) return;
  const float* r0 = Wfus + (size_t)wid * n;
  const float* r1 = Wfi  + (size_t)wid * n;
  const float* r2 = Wfq  + (size_t)wid * n;
  float a0 = 0.f, a1 = 0.f, a2 = 0.f;
  for (int k = lane; k < n; k += 32) { a0 += r0[k] * h[k]; a1 += r1[k] * ci[k]; a2 += r2[k] * cq[k]; }
  a0 = wave_sum(a0); a1 = wave_sum(a1); a2 = wave_sum(a2);
  if (lane == 0) fs[wid] = tanhf(a0 + bfus[wid] + mw[0] * a1 + mw[1] * a2);
}

// ---------------------------------------------------------------- decoder gates g[r] = Wi[r]·fs + Wh[r]·h + b[r]
__global__ __launch_bounds__(256) void dec_gates(const float* __restrict__ Wi, const float* __restrict__ Wh,
                                                 const float* __restrict__ b, const float* __restrict__ fs,
                                                 const float* __restrict__ h, float* __restrict__ g,
                                                 int rows, int n) {
  const int wid = (blockIdx.x * blockDim.x + threadIdx.x) >> 5;
  const int lane = threadIdx.x & 31;
  if (wid >= rows) return;
  const float* r0 = Wi + (size_t)wid * n;
  const float* r1 = Wh + (size_t)wid * n;
  float a0 = 0.f, a1 = 0.f;
  for (int k = lane; k < n; k += 32) { a0 += r0[k] * fs[k]; a1 += r1[k] * h[k]; }
  float s = wave_sum(a0 + a1);
  if (lane == 0) g[wid] = s + b[wid];
}

__global__ __launch_bounds__(256) void dec_update(const float* __restrict__ g, float* __restrict__ h,
                                                  float* __restrict__ c, int H_) {
  int j = blockIdx.x * blockDim.x + threadIdx.x;
  if (j >= H_) return;
  float ig = g[j], fg = g[H_ + j], gg = g[2 * H_ + j], og = g[3 * H_ + j];
  float cc = sigmoidf(fg) * c[j] + sigmoidf(ig) * tanhf(gg);
  c[j] = cc;
  h[j] = sigmoidf(og) * tanhf(cc);
}

// ---------------------------------------------------------------- logits with bf16 weights (L2-resident)
__global__ __launch_bounds__(256) void logits_bf16(const u16* __restrict__ Wf, const float* __restrict__ bias,
                                                   const float* __restrict__ h, float* __restrict__ out,
                                                   int rows, int n) {
  const int wid = (blockIdx.x * blockDim.x + threadIdx.x) >> 5;
  const int lane = threadIdx.x & 31;
  if (wid >= rows) return;
  const u16* row = Wf + (size_t)wid * n;
  float acc = 0.f;
  for (int k = lane * 2; k < n; k += 64) {
    unsigned u = *(const unsigned*)(row + k);
    acc += bf2f((u16)(u & 0xffffu)) * h[k] + bf2f((u16)(u >> 16)) * h[k + 1];
  }
  acc = wave_sum(acc);
  if (lane == 0) out[wid] = acc + bias[wid];
}

// ================================================================ host
extern "C" void kernel_launch(void* const* d_in, const int* in_sizes, int n_in,
                              void* d_out, int out_size, void* d_ws, size_t ws_size,
                              hipStream_t stream) {
  (void)in_sizes; (void)n_in; (void)out_size; (void)ws_size;
  constexpr int O_ = 1024, EI = 512, EQ = 512, DI = 2048, DQ = 300, DQp = 320;
  constexpr int LI_ = 1024, LQ_ = 32, V_ = 32000, T_ = 20;

  const float* img_feats = (const float*)d_in[0];
  const float* q_feats   = (const float*)d_in[1];
  const float* Wi_imgf = (const float*)d_in[2];  const float* Wh_imgf = (const float*)d_in[3];
  const float* b_imgf  = (const float*)d_in[4];
  const float* Wi_imgb = (const float*)d_in[5];  const float* Wh_imgb = (const float*)d_in[6];
  const float* b_imgb  = (const float*)d_in[7];
  const float* Wi_qf = (const float*)d_in[8];    const float* Wh_qf = (const float*)d_in[9];
  const float* b_qf  = (const float*)d_in[10];
  const float* Wi_qb = (const float*)d_in[11];   const float* Wh_qb = (const float*)d_in[12];
  const float* b_qb  = (const float*)d_in[13];
  const float* Wimg_h = (const float*)d_in[14];  const float* Wimg_e = (const float*)d_in[15];
  const float* bimg   = (const float*)d_in[16];
  const float* Wq_h = (const float*)d_in[17];    const float* Wq_e = (const float*)d_in[18];
  const float* bq   = (const float*)d_in[19];
  const float* Wmod = (const float*)d_in[20];    const float* bmod = (const float*)d_in[21];
  const float* Wmod_img = (const float*)d_in[22]; const float* Wmod_q = (const float*)d_in[23];
  const float* Wfus_img = (const float*)d_in[24]; const float* Wfus_q = (const float*)d_in[25];
  const float* Wfus = (const float*)d_in[26];    const float* bfus = (const float*)d_in[27];
  const float* Wi_dec = (const float*)d_in[28];  const float* Wh_dec = (const float*)d_in[29];
  const float* b_dec  = (const float*)d_in[30];
  const float* Wfin = (const float*)d_in[31];    const float* bfin = (const float*)d_in[32];
  float* out = (float*)d_out;

  // workspace carve (256B aligned)
  size_t off = 0;
  auto alloc = [&](size_t bytes) -> void* {
    off = (off + 255) & ~(size_t)255;
    void* p = (char*)d_ws + off;
    off += bytes;
    return p;
  };
  u16* img_bf  = (u16*)alloc((size_t)LI_ * DI * 2);
  u16* WiF_bf  = (u16*)alloc((size_t)4 * EI * DI * 2);
  u16* WiB_bf  = (u16*)alloc((size_t)4 * EI * DI * 2);
  u16* q_bf    = (u16*)alloc((size_t)LQ_ * DQp * 2);
  u16* WiQF_bf = (u16*)alloc((size_t)4 * EQ * DQp * 2);
  u16* WiQB_bf = (u16*)alloc((size_t)4 * EQ * DQp * 2);
  u16* WhIF_bf = (u16*)alloc((size_t)4 * EI * EI * 2);
  u16* WhIB_bf = (u16*)alloc((size_t)4 * EI * EI * 2);
  u16* WhQF_bf = (u16*)alloc((size_t)4 * EQ * EQ * 2);
  u16* WhQB_bf = (u16*)alloc((size_t)4 * EQ * EQ * 2);
  u16* Wfin_bf = (u16*)alloc((size_t)V_ * O_ * 2);
  float* xpIF   = (float*)alloc((size_t)LI_ * 4 * EI * 4);
  float* xpIB   = (float*)alloc((size_t)LI_ * 4 * EI * 4);
  float* xpQF   = (float*)alloc((size_t)LQ_ * 4 * EQ * 4);
  float* xpQB   = (float*)alloc((size_t)LQ_ * 4 * EQ * 4);
  float* img_emb = (float*)alloc((size_t)LI_ * 2 * EI * 4);
  float* q_emb   = (float*)alloc((size_t)LQ_ * 2 * EQ * 4);
  float* e_img = (float*)alloc(LI_ * 4);
  float* e_q   = (float*)alloc(LQ_ * 4);
  float* ctx_img = (float*)alloc(2 * EI * 4);
  float* ctx_q   = (float*)alloc(2 * EQ * 4);
  float* mw = (float*)alloc(2 * 4);
  float* fs = (float*)alloc(O_ * 4);
  float* g  = (float*)alloc(4 * O_ * 4);
  float* hc = (float*)alloc(2 * O_ * 4);
  float* h = hc, * c = hc + O_;

  // ---- bf16 conversions (once)
  auto cvt = [&](const float* s, u16* d, int rows, int K, int Kp) {
    size_t tot = (size_t)rows * Kp;
    cvt_pad<<<dim3((unsigned)((tot + 255) / 256)), 256, 0, stream>>>(s, d, rows, K, Kp);
  };
  cvt(img_feats, img_bf, LI_, DI, DI);
  cvt(Wi_imgf, WiF_bf, 4 * EI, DI, DI);
  cvt(Wi_imgb, WiB_bf, 4 * EI, DI, DI);
  cvt(q_feats, q_bf, LQ_, DQ, DQp);
  cvt(Wi_qf, WiQF_bf, 4 * EQ, DQ, DQp);
  cvt(Wi_qb, WiQB_bf, 4 * EQ, DQ, DQp);
  cvt(Wh_imgf, WhIF_bf, 4 * EI, EI, EI);
  cvt(Wh_imgb, WhIB_bf, 4 * EI, EI, EI);
  cvt(Wh_qf, WhQF_bf, 4 * EQ, EQ, EQ);
  cvt(Wh_qb, WhQB_bf, 4 * EQ, EQ, EQ);
  cvt(Wfin, Wfin_bf, V_, O_, O_);

  // ---- input-projection GEMMs on WMMA
  dim3 gi((LI_ + 63) / 64, (4 * EI) / 128);
  wmma_gemm_bf16<<<gi, 256, 0, stream>>>(img_bf, WiF_bf, xpIF, LI_, 4 * EI, DI);
  wmma_gemm_bf16<<<gi, 256, 0, stream>>>(img_bf, WiB_bf, xpIB, LI_, 4 * EI, DI);
  dim3 gq((LQ_ + 63) / 64, (4 * EQ) / 128);
  wmma_gemm_bf16<<<gq, 256, 0, stream>>>(q_bf, WiQF_bf, xpQF, LQ_, 4 * EQ, DQp);
  wmma_gemm_bf16<<<gq, 256, 0, stream>>>(q_bf, WiQB_bf, xpQB, LQ_, 4 * EQ, DQp);

  // ---- 4 concurrent persistent LSTM scans (img fwd/bwd, q fwd/bwd)
  lstm_scan4<<<4, 1024, 0, stream>>>(xpIF, xpIB, xpQF, xpQB,
                                     WhIF_bf, WhIB_bf, WhQF_bf, WhQB_bf,
                                     b_imgf, b_imgb, b_qf, b_qb,
                                     img_emb, q_emb, LI_, LQ_);

  // ---- static encoder attention scores
  rowdot<<<(LI_ * 32 + 255) / 256, 256, 0, stream>>>(img_emb, Wimg_e, bimg, 1, e_img, LI_, 2 * EI);
  rowdot<<<(LQ_ * 32 + 255) / 256, 256, 0, stream>>>(q_emb, Wq_e, bq, 1, e_q, LQ_, 2 * EQ);

  // ---- decoder
  zero_f32<<<(2 * O_ + 255) / 256, 256, 0, stream>>>(hc, 2 * O_);
  for (int t = 0; t < T_; ++t) {
    attn_ctx<<<1, 1024, 0, stream>>>(Wimg_h, h, e_img, img_emb, ctx_img, LI_, 2 * EI, O_);
    attn_ctx<<<1, 1024, 0, stream>>>(Wq_h, h, e_q, q_emb, ctx_q, LQ_, 2 * EQ, O_);
    mod_weights<<<1, 256, 0, stream>>>(Wmod, bmod, Wmod_img, Wmod_q, h, ctx_img, ctx_q, mw, O_);
    fuse_fs<<<(O_ * 32 + 255) / 256, 256, 0, stream>>>(Wfus, bfus, Wfus_img, Wfus_q,
                                                       h, ctx_img, ctx_q, mw, fs, O_, O_);
    dec_gates<<<(4 * O_ * 32 + 255) / 256, 256, 0, stream>>>(Wi_dec, Wh_dec, b_dec, fs, h, g, 4 * O_, O_);
    dec_update<<<(O_ + 255) / 256, 256, 0, stream>>>(g, h, c, O_);
    logits_bf16<<<(V_ * 32 + 255) / 256, 256, 0, stream>>>(Wfin_bf, bfin, h, out + (size_t)t * V_, V_, O_);
  }
}
